// DIMKT_74337293959382
// MI455X (gfx1250) — compile-verified, matrix operand order checked by reading, not run on hardware
//
#include <hip/hip_runtime.h>
#include <hip/hip_bf16.h>
#include <math.h>

#define B_ 512
#define S_ 200
#define D_ 512

typedef __attribute__((ext_vector_type(16))) __bf16 v16bf;
typedef __attribute__((ext_vector_type(8)))  float  v8f;
typedef unsigned u32x4 __attribute__((ext_vector_type(4)));
typedef int      i32x8 __attribute__((ext_vector_type(8)));
typedef int      i32x4 __attribute__((ext_vector_type(4)));

#define BM 128
#define BN 128
#define BK 32

__device__ inline __bf16 f2bf(float f) {
  union { float f; unsigned u; } in; in.f = f;
  unsigned r = in.u + 0x7FFFu + ((in.u >> 16) & 1u);   // round-to-nearest-even
  union { unsigned short s; __bf16 b; } out; out.s = (unsigned short)(r >> 16);
  return out.b;
}

__device__ inline float sigm(float x) { return 1.0f / (1.0f + __expf(-x)); }

// ---- Tensor Data Mover: 2-D tile (BK elems x BM rows, bf16) global -> LDS ----
// D# per CDNA5 ISA: group0 {count,lds_addr,global_addr,type=2},
// group1 {data_size=2B, tensor_dim0/1, tile_dim0=BK, tile_dim1=BM, dim0_stride}.
// This toolchain exposes the 6-arg builtin: (g0, g1, g2, g3, g4, cpol).
__device__ inline void tdm_load_2d(unsigned lds_byte_off, const void* gptr,
                                   int tensor_rows, int tensor_elems,
                                   int stride_elems) {
  unsigned long long ga = (unsigned long long)(size_t)gptr;
  u32x4 g0;
  g0[0] = 1u;                                   // count = 1 (valid descriptor)
  g0[1] = lds_byte_off;                         // lds_addr
  g0[2] = (unsigned)ga;                         // global_addr[31:0]
  g0[3] = (unsigned)(ga >> 32) | (2u << 30);    // global_addr[56:32] | type=2
  i32x8 g1;
  g1[0] = 1 << 16;                              // data_size = 1 -> 2 bytes
  g1[1] = (tensor_elems & 0xFFFF) << 16;        // tensor_dim0[15:0]
  g1[2] = ((tensor_elems >> 16) & 0xFFFF) | ((tensor_rows & 0xFFFF) << 16);
  g1[3] = ((tensor_rows >> 16) & 0xFFFF) | (BK << 16);   // tile_dim0 = BK
  g1[4] = BM;                                   // tile_dim1 = BM, tile_dim2 = 0
  g1[5] = stride_elems;                         // tensor_dim0_stride[31:0]
  g1[6] = 0;                                    // stride[47:32] | dim1_stride lo
  g1[7] = 0;
  i32x4 gz = {0, 0, 0, 0};
  i32x8 gz8 = {0, 0, 0, 0, 0, 0, 0, 0};
  __builtin_amdgcn_tensor_load_to_lds(g0, g1, gz, gz, gz8, 0);
}

// A/B 16-bit fragment from an LDS tile stored row-major [rows][BK].
// Per ISA 7.12.2: lanes 0-15 -> row=rbase+lane, K {0..7,16..23}; lanes 16-31 -> K {8..15,24..31}.
__device__ inline v16bf load_frag(const __bf16* tile, int ld, int rbase, int lane) {
  const int r = rbase + (lane & 15);
  const __bf16* p = tile + r * ld + ((lane & 16) ? 8 : 0);
  union { uint4 q[2]; v16bf v; } f;
  f.q[0] = *(const uint4*)(p);
  f.q[1] = *(const uint4*)(p + 16);
  return f.v;
}

__device__ inline v8f wmma_bf16(v16bf a, v16bf b, v8f c) {
  return __builtin_amdgcn_wmma_f32_16x16x32_bf16(false, a, false, b, (short)0, c,
                                                 false, false);
}

// Core 128x128 tile GEMM, TDM-fed + LDS double-buffered.
// C = A[rowBase:+128, :K] * W^T (W stored [N][K] bf16). DUAL shares A for W1,W2.
// sA/sB1/sB2 are 2*128*32 bf16 (double buffers).
template<bool DUAL>
__device__ inline void gemm_core(const __bf16* __restrict__ A, int lda,
                                 const __bf16* __restrict__ W1,
                                 const __bf16* __restrict__ W2, int ldw, int K,
                                 int rowBase, int colBase,
                                 __bf16* sA, __bf16* sB1, __bf16* sB2,
                                 v8f acc1[4][2], v8f acc2[4][2]) {
  const int tid  = threadIdx.x;
  const int lane = tid & 31;
  const int wave = tid >> 5;
  const int wm = wave >> 2;            // 0..1 : 64 rows each
  const int wn = wave & 3;             // 0..3 : 32 cols each
  const v8f vz = {};
  for (int m = 0; m < 4; ++m)
    for (int n = 0; n < 2; ++n) { acc1[m][n] = vz; if (DUAL) acc2[m][n] = vz; }

  const unsigned offA  = (unsigned)(size_t)sA;
  const unsigned offB1 = (unsigned)(size_t)sB1;
  const unsigned offB2 = (unsigned)(size_t)sB2;
  const int iters = K / BK;

  auto issue = [&](int it) {
    const int k0 = it * BK;
    const unsigned bo = (unsigned)(it & 1) * (BM * BK * 2);  // byte offset of buffer
    tdm_load_2d(offA + bo,  A  + (size_t)rowBase * lda + k0, BM, K - k0, lda);
    tdm_load_2d(offB1 + bo, W1 + (size_t)colBase * ldw + k0, BN, K - k0, ldw);
    if (DUAL)
      tdm_load_2d(offB2 + bo, W2 + (size_t)colBase * ldw + k0, BN, K - k0, ldw);
  };

  if (tid == 0) issue(0);
  for (int it = 0; it < iters; ++it) {
    const bool pf = (it + 1 < iters);
    if (tid == 0 && pf) issue(it + 1);           // prefetch next K-slab into alt buffer
    if (pf) {
      // allow the just-issued prefetch set to stay in flight (TDM completes in order)
      __builtin_amdgcn_s_wait_tensorcnt(DUAL ? 3 : 2);
    } else {
      __builtin_amdgcn_s_wait_tensorcnt(0);
    }
    __syncthreads();
    const int eo = (it & 1) * (BM * BK);
    v16bf af[4], bf1[2], bf2[2];
    #pragma unroll
    for (int m = 0; m < 4; ++m) af[m] = load_frag(sA + eo, BK, wm * 64 + m * 16, lane);
    #pragma unroll
    for (int n = 0; n < 2; ++n) {
      bf1[n] = load_frag(sB1 + eo, BK, wn * 32 + n * 16, lane);
      if (DUAL) bf2[n] = load_frag(sB2 + eo, BK, wn * 32 + n * 16, lane);
    }
    #pragma unroll
    for (int m = 0; m < 4; ++m)
      #pragma unroll
      for (int n = 0; n < 2; ++n) {
        acc1[m][n] = wmma_bf16(af[m], bf1[n], acc1[m][n]);
        if (DUAL) acc2[m][n] = wmma_bf16(af[m], bf2[n], acc2[m][n]);
      }
    __syncthreads();
  }
}

// ---- weight convert + transpose: W f32 [K][N] -> Wt bf16 [N][K] ----
__global__ __launch_bounds__(256) void transpose_w_kernel(
    const float* __restrict__ W, __bf16* __restrict__ Wt, int K, int N) {
  int i = blockIdx.x * 256 + threadIdx.x;
  if (i >= K * N) return;
  int n = i / K, k = i - n * K;
  Wt[(size_t)n * K + k] = f2bf(W[(size_t)k * N + n]);
}

// ---- x_all = concat(qe,ce,qde,cde) @ W_x + b_x ; output f32 time-major [S][B][D] ----
// A operand is gathered + converted per K-slab (not TDM-able); W tile arrives via TDM.
__global__ __launch_bounds__(256) void gemm_x_kernel(
    const int* __restrict__ qseq, const int* __restrict__ cseq,
    const int* __restrict__ qdseq, const int* __restrict__ cdseq,
    const float* __restrict__ Eq, const float* __restrict__ Ec,
    const float* __restrict__ Eqd, const float* __restrict__ Ecd,
    const __bf16* __restrict__ Wxt, const float* __restrict__ bx,
    float* __restrict__ x_all) {
  __shared__ __attribute__((aligned(16))) __bf16 sA[BM * BK];
  __shared__ __attribute__((aligned(16))) __bf16 sB[BN * BK];
  const int tid = threadIdx.x, lane = tid & 31, wave = tid >> 5;
  const int wm = wave >> 2, wn = wave & 3;
  const int rowBase = blockIdx.x * BM, colBase = blockIdx.y * BN;
  const int ldr = tid >> 1, lko = (tid & 1) * 16;
  const int r = rowBase + ldr;
  const int b = r & (B_ - 1);          // row layout: r = s*B + b
  const int s = r >> 9;
  const unsigned offB = (unsigned)(size_t)sB;
  const v8f vz = {};
  v8f acc[4][2];
  for (int m = 0; m < 4; ++m) for (int n = 0; n < 2; ++n) acc[m][n] = vz;
  for (int k0 = 0; k0 < 4 * D_; k0 += BK) {
    if (tid == 0)
      tdm_load_2d(offB, Wxt + (size_t)colBase * (4 * D_) + k0, BN, 4 * D_ - k0, 4 * D_);
    const int tbl = k0 >> 9;           // each 512-wide chunk = one embedding table
    const int* seq = (tbl == 0) ? qseq : (tbl == 1) ? cseq : (tbl == 2) ? qdseq : cdseq;
    const float* E = (tbl == 0) ? Eq : (tbl == 1) ? Ec : (tbl == 2) ? Eqd : Ecd;
    const int idx = seq[b * S_ + s];
    const float* src = E + (size_t)idx * D_ + (k0 & (D_ - 1)) + lko;
    __bf16* dst = sA + ldr * BK + lko;
    #pragma unroll
    for (int j = 0; j < 4; ++j) {
      float4 v = ((const float4*)src)[j];
      dst[4 * j + 0] = f2bf(v.x); dst[4 * j + 1] = f2bf(v.y);
      dst[4 * j + 2] = f2bf(v.z); dst[4 * j + 3] = f2bf(v.w);
    }
    __builtin_amdgcn_s_wait_tensorcnt(0);
    __syncthreads();
    v16bf af[4], bfr[2];
    #pragma unroll
    for (int m = 0; m < 4; ++m) af[m] = load_frag(sA, BK, wm * 64 + m * 16, lane);
    #pragma unroll
    for (int n = 0; n < 2; ++n) bfr[n] = load_frag(sB, BK, wn * 32 + n * 16, lane);
    #pragma unroll
    for (int m = 0; m < 4; ++m)
      #pragma unroll
      for (int n = 0; n < 2; ++n) acc[m][n] = wmma_bf16(af[m], bfr[n], acc[m][n]);
    __syncthreads();
  }
  for (int m = 0; m < 4; ++m)
    for (int n = 0; n < 2; ++n)
      for (int e = 0; e < 8; ++e) {
        int row = rowBase + wm * 64 + m * 16 + e + ((lane & 16) ? 8 : 0);
        int col = colBase + wn * 32 + n * 16 + (lane & 15);
        x_all[(size_t)row * D_ + col] = acc[m][n][e] + bx[col];
      }
}

// ---- init t=0: h, d_bf16, ki_in, pka_in[:,D:] ----
__global__ __launch_bounds__(256) void k0_init_kernel(
    const float* __restrict__ x_all, const float* __restrict__ h0,
    const int* __restrict__ corr_seq, const int* __restrict__ qd_seq,
    const int* __restrict__ cd_seq, const float* __restrict__ Ecorr,
    const float* __restrict__ Eqd, const float* __restrict__ Ecd,
    float* __restrict__ h, __bf16* __restrict__ dbf,
    __bf16* __restrict__ ki, __bf16* __restrict__ pkain) {
  int i = blockIdx.x * 256 + threadIdx.x;
  if (i >= B_ * D_) return;
  int b = i >> 9, col = i & (D_ - 1);
  float hp = h0[i];
  h[i] = hp;
  dbf[i] = f2bf(x_all[i] - hp);                 // x_t at t=0 is slice 0
  int ic = corr_seq[b * S_], iq = qd_seq[b * S_], id = cd_seq[b * S_];
  size_t kb = (size_t)b * (4 * D_);
  ki[kb + col] = f2bf(hp);
  float ec = Ecorr[ic * D_ + col], eq = Eqd[iq * D_ + col], ed = Ecd[id * D_ + col];
  ki[kb + D_ + col] = f2bf(ec);
  ki[kb + 2 * D_ + col] = f2bf(eq);
  ki[kb + 3 * D_ + col] = f2bf(ed);
  pkain[(size_t)b * (2 * D_) + D_ + col] = f2bf(ec);
}

// ---- gamma = sigmoid(ki_in @ W_ki + b_ki), K=2048 ----
__global__ __launch_bounds__(256) void gemm_gamma_kernel(
    const __bf16* __restrict__ ki, const __bf16* __restrict__ Wkit,
    const float* __restrict__ bki, float* __restrict__ gamma) {
  __shared__ __attribute__((aligned(16))) __bf16 sA[2 * BM * BK];
  __shared__ __attribute__((aligned(16))) __bf16 sB1[2 * BN * BK];
  v8f a1[4][2], a2[4][2];
  const int rowBase = blockIdx.x * BM, colBase = blockIdx.y * BN;
  gemm_core<false>(ki, 4 * D_, Wkit, nullptr, 4 * D_, 4 * D_, rowBase, colBase,
                   sA, sB1, sB1, a1, a2);
  const int lane = threadIdx.x & 31, wave = threadIdx.x >> 5;
  const int wm = wave >> 2, wn = wave & 3;
  for (int m = 0; m < 4; ++m)
    for (int n = 0; n < 2; ++n)
      for (int e = 0; e < 8; ++e) {
        int row = rowBase + wm * 64 + m * 16 + e + ((lane & 16) ? 8 : 0);
        int col = colBase + wn * 32 + n * 16 + (lane & 15);
        gamma[(size_t)row * D_ + col] = sigm(a1[m][n][e] + bki[col]);
      }
}

// ---- sdf = sigmoid(d@W_s1+b1) * tanh(d@W_s2+b2) -> bf16 into pka_in[:, :D] ----
__global__ __launch_bounds__(256) void gemm_sdf_kernel(
    const __bf16* __restrict__ dbf, const __bf16* __restrict__ W1t,
    const __bf16* __restrict__ W2t, const float* __restrict__ b1,
    const float* __restrict__ b2, __bf16* __restrict__ pkain) {
  __shared__ __attribute__((aligned(16))) __bf16 sA[2 * BM * BK];
  __shared__ __attribute__((aligned(16))) __bf16 sB1[2 * BN * BK];
  __shared__ __attribute__((aligned(16))) __bf16 sB2[2 * BN * BK];
  v8f a1[4][2], a2[4][2];
  const int rowBase = blockIdx.x * BM, colBase = blockIdx.y * BN;
  gemm_core<true>(dbf, D_, W1t, W2t, D_, D_, rowBase, colBase, sA, sB1, sB2, a1, a2);
  const int lane = threadIdx.x & 31, wave = threadIdx.x >> 5;
  const int wm = wave >> 2, wn = wave & 3;
  for (int m = 0; m < 4; ++m)
    for (int n = 0; n < 2; ++n)
      for (int e = 0; e < 8; ++e) {
        int row = rowBase + wm * 64 + m * 16 + e + ((lane & 16) ? 8 : 0);
        int col = colBase + wn * 32 + n * 16 + (lane & 15);
        float sg = sigm(a1[m][n][e] + b1[col]);
        float th = tanhf(a2[m][n][e] + b2[col]);
        pkain[(size_t)row * (2 * D_) + col] = f2bf(sg * th);
      }
}

// ---- pka dual-GEMM (K=1024) fused with h-update, y row-dot, next-step build ----
__global__ __launch_bounds__(256) void gemm_pka_kernel(
    const __bf16* __restrict__ pkain, __bf16* __restrict__ pkain_w,
    const __bf16* __restrict__ Wp1t, const __bf16* __restrict__ Wp2t,
    const float* __restrict__ bp1, const float* __restrict__ bp2,
    const float* __restrict__ gamma, float* __restrict__ h,
    const float* __restrict__ xnext, float* __restrict__ y_acc,
    __bf16* __restrict__ dbf, __bf16* __restrict__ ki,
    const int* __restrict__ corr_seq, const int* __restrict__ qd_seq,
    const int* __restrict__ cd_seq, const float* __restrict__ Ecorr,
    const float* __restrict__ Eqd, const float* __restrict__ Ecd,
    int t, int build_next) {
  __shared__ __attribute__((aligned(16))) __bf16 sA[2 * BM * BK];
  __shared__ __attribute__((aligned(16))) __bf16 sB1[2 * BN * BK];
  __shared__ __attribute__((aligned(16))) __bf16 sB2[2 * BN * BK];
  __shared__ float yred[BM];
  if (threadIdx.x < BM) yred[threadIdx.x] = 0.0f;
  v8f a1[4][2], a2[4][2];
  const int rowBase = blockIdx.x * BM, colBase = blockIdx.y * BN;
  gemm_core<true>(pkain, 2 * D_, Wp1t, Wp2t, 2 * D_, 2 * D_, rowBase, colBase,
                  sA, sB1, sB2, a1, a2);
  const int lane = threadIdx.x & 31, wave = threadIdx.x >> 5;
  const int wm = wave >> 2, wn = wave & 3;
  for (int m = 0; m < 4; ++m)
    for (int n = 0; n < 2; ++n)
      for (int e = 0; e < 8; ++e) {
        int row = rowBase + wm * 64 + m * 16 + e + ((lane & 16) ? 8 : 0);
        int col = colBase + wn * 32 + n * 16 + (lane & 15);
        float pka = sigm(a1[m][n][e] + bp1[col]) * tanhf(a2[m][n][e] + bp2[col]);
        size_t ix = (size_t)row * D_ + col;
        float g = gamma[ix];
        float hp = h[ix];
        float hn = g * hp + (1.0f - g) * pka;
        h[ix] = hn;
        float xn = xnext[ix];
        atomicAdd(&yred[row - rowBase], xn * hn);          // LDS f32 atomic
        dbf[ix] = f2bf(xn - hn);                           // d for step t+1
        size_t kb = (size_t)row * (4 * D_);
        ki[kb + col] = f2bf(hn);                           // h chunk of ki_in
        if (build_next) {
          int tn = t + 1;
          int ic = corr_seq[row * S_ + tn];
          int iq = qd_seq[row * S_ + tn];
          int id = cd_seq[row * S_ + tn];
          float ec = Ecorr[ic * D_ + col];
          ki[kb + D_ + col]     = f2bf(ec);
          ki[kb + 2 * D_ + col] = f2bf(Eqd[iq * D_ + col]);
          ki[kb + 3 * D_ + col] = f2bf(Ecd[id * D_ + col]);
          pkain_w[(size_t)row * (2 * D_) + D_ + col] = f2bf(ec);
        }
      }
  __syncthreads();
  if (threadIdx.x < BM)
    atomicAdd(&y_acc[(size_t)t * B_ + rowBase + threadIdx.x], yred[threadIdx.x]);
}

__global__ __launch_bounds__(256) void zero_f32_kernel(float* __restrict__ p, int n) {
  int i = blockIdx.x * 256 + threadIdx.x;
  if (i < n) p[i] = 0.0f;
}

__global__ __launch_bounds__(256) void final_kernel(const float* __restrict__ y_acc,
                                                    float* __restrict__ out) {
  int i = blockIdx.x * 256 + threadIdx.x;   // over B*S
  if (i >= B_ * S_) return;
  int b = i / S_, s = i - b * S_;
  out[i] = (s < S_ - 1) ? sigm(y_acc[(size_t)s * B_ + b]) : 0.0f;
}

extern "C" void kernel_launch(void* const* d_in, const int* in_sizes, int n_in,
                              void* d_out, int out_size, void* d_ws, size_t ws_size,
                              hipStream_t stream) {
  (void)in_sizes; (void)n_in; (void)out_size; (void)ws_size;
  const int* qseq  = (const int*)d_in[0];
  const int* cseq  = (const int*)d_in[1];
  const int* qdseq = (const int*)d_in[2];
  const int* cdseq = (const int*)d_in[3];
  const int* crseq = (const int*)d_in[4];
  const float* Eq  = (const float*)d_in[5];
  const float* Ec  = (const float*)d_in[6];
  const float* Eqd = (const float*)d_in[7];
  const float* Ecd = (const float*)d_in[8];
  const float* Ecr = (const float*)d_in[9];
  const float* Wx  = (const float*)d_in[10]; const float* bx  = (const float*)d_in[11];
  const float* Ws1 = (const float*)d_in[12]; const float* bs1 = (const float*)d_in[13];
  const float* Ws2 = (const float*)d_in[14]; const float* bs2 = (const float*)d_in[15];
  const float* Wp1 = (const float*)d_in[16]; const float* bp1 = (const float*)d_in[17];
  const float* Wp2 = (const float*)d_in[18]; const float* bp2 = (const float*)d_in[19];
  const float* Wki = (const float*)d_in[20]; const float* bki = (const float*)d_in[21];
  const float* h0  = (const float*)d_in[22];
  float* out = (float*)d_out;

  char* p = (char*)d_ws;
  float*  x_all = (float*)p;  p += (size_t)S_ * B_ * D_ * 4;      // 200 MB, time-major
  __bf16* Wxt   = (__bf16*)p; p += (size_t)D_ * 4 * D_ * 2;
  __bf16* Ws1t  = (__bf16*)p; p += (size_t)D_ * D_ * 2;
  __bf16* Ws2t  = (__bf16*)p; p += (size_t)D_ * D_ * 2;
  __bf16* Wp1t  = (__bf16*)p; p += (size_t)D_ * 2 * D_ * 2;
  __bf16* Wp2t  = (__bf16*)p; p += (size_t)D_ * 2 * D_ * 2;
  __bf16* Wkit  = (__bf16*)p; p += (size_t)D_ * 4 * D_ * 2;
  __bf16* dbf   = (__bf16*)p; p += (size_t)B_ * D_ * 2;
  __bf16* ki    = (__bf16*)p; p += (size_t)B_ * 4 * D_ * 2;
  __bf16* pkain = (__bf16*)p; p += (size_t)B_ * 2 * D_ * 2;
  float*  gamma = (float*)p;  p += (size_t)B_ * D_ * 4;
  float*  hbuf  = (float*)p;  p += (size_t)B_ * D_ * 4;
  float*  y_acc = (float*)p;  p += (size_t)S_ * B_ * 4;

  auto tgrid = [](int k, int n) { return dim3((unsigned)((k * n + 255) / 256)); };
  transpose_w_kernel<<<tgrid(4 * D_, D_), 256, 0, stream>>>(Wx,  Wxt,  4 * D_, D_);
  transpose_w_kernel<<<tgrid(D_, D_),     256, 0, stream>>>(Ws1, Ws1t, D_, D_);
  transpose_w_kernel<<<tgrid(D_, D_),     256, 0, stream>>>(Ws2, Ws2t, D_, D_);
  transpose_w_kernel<<<tgrid(2 * D_, D_), 256, 0, stream>>>(Wp1, Wp1t, 2 * D_, D_);
  transpose_w_kernel<<<tgrid(2 * D_, D_), 256, 0, stream>>>(Wp2, Wp2t, 2 * D_, D_);
  transpose_w_kernel<<<tgrid(4 * D_, D_), 256, 0, stream>>>(Wki, Wkit, 4 * D_, D_);

  // Big GEMM: M = S*B = 102400 rows, K = 2048, N = 512
  gemm_x_kernel<<<dim3((S_ * B_) / BM, D_ / BN), 256, 0, stream>>>(
      qseq, cseq, qdseq, cdseq, Eq, Ec, Eqd, Ecd, Wxt, bx, x_all);

  zero_f32_kernel<<<dim3((S_ * B_ + 255) / 256), 256, 0, stream>>>(y_acc, S_ * B_);
  k0_init_kernel<<<dim3((B_ * D_ + 255) / 256), 256, 0, stream>>>(
      x_all, h0, crseq, qdseq, cdseq, Ecr, Eqd, Ecd, hbuf, dbf, ki, pkain);

  for (int t = 0; t < S_ - 1; ++t) {
    gemm_gamma_kernel<<<dim3(B_ / BM, D_ / BN), 256, 0, stream>>>(ki, Wkit, bki, gamma);
    gemm_sdf_kernel<<<dim3(B_ / BM, D_ / BN), 256, 0, stream>>>(
        dbf, Ws1t, Ws2t, bs1, bs2, pkain);
    const float* xnext = x_all + (size_t)(t + 1) * B_ * D_;
    gemm_pka_kernel<<<dim3(B_ / BM, D_ / BN), 256, 0, stream>>>(
        pkain, pkain, Wp1t, Wp2t, bp1, bp2, gamma, hbuf, xnext, y_acc, dbf, ki,
        crseq, qdseq, cdseq, Ecr, Eqd, Ecd, t, (t < S_ - 2) ? 1 : 0);
  }

  final_kernel<<<dim3((B_ * S_ + 255) / 256), 256, 0, stream>>>(y_acc, out);
}
